// DiffAttention2D_2989297238714
// MI455X (gfx1250) — compile-verified, matrix operand order checked by reading, not run on hardware
//
#include <hip/hip_runtime.h>

// ---------------------------------------------------------------------------
// DiffAttention2D for MI455X (gfx1250, wave32, WMMA).
// All GEMM stages use v_wmma_f32_16x16x32_bf16 with f32 accumulate and 2x2
// register blocking (32x32 per wave, 4 accumulators, operand reuse).
// k_pv stages the shared v-tile through LDS with GLOBAL_LOAD_ASYNC_TO_LDS_B128
// via inline asm (ASYNCcnt-tracked, ISA §15.18.3 opcode 98).
// B=2, C=256, heads=8, d=32, N=48*48=2304. d==K(32) -> one WMMA per score tile.
// ---------------------------------------------------------------------------

#define Bn   2
#define Cc   256
#define Hh   8
#define Dd   32
#define Nn   2304          // 48*48
#define NT   144           // Nn/16 tiles
#define SCALE 0.17677669529663687f   // 1/sqrt(32)

typedef __attribute__((ext_vector_type(16))) __bf16        bf16x16;
typedef __attribute__((ext_vector_type(8)))  float         f32x8;
typedef __attribute__((ext_vector_type(4)))  unsigned int  u32x4;

#if defined(__gfx1250__)
#define HAVE_ASYNC_LDS 1
#else
#define HAVE_ASYNC_LDS 0
#endif

// ---- WMMA wrapper ---------------------------------------------------------
__device__ __forceinline__ f32x8 wmma_bf16(bf16x16 a, bf16x16 b, f32x8 c) {
  return __builtin_amdgcn_wmma_f32_16x16x32_bf16(false, a, false, b,
                                                 (short)0, c, false, false);
}

// ---- 16x32 operand tile load (A: rows x K; B via K-contiguous storage) ----
// ISA lane layout: lanes 0-15 carry row/col r=lane, elems 0..7 -> K+0..7,
// elems 8..15 -> K+16..23; lanes 16-31 get K+8..15 / K+24..31.
// Two 16B vector loads per lane (all strides are 16B-aligned by construction).
__device__ __forceinline__ bf16x16 load_tile(const __bf16* base, int ld,
                                             int rc0, int k0) {
  int lane = threadIdx.x & 31;
  const __bf16* p = base + (size_t)(rc0 + (lane & 15)) * ld
                         + k0 + ((lane >> 4) << 3);
  union { u32x4 u[2]; bf16x16 v; } t;
  t.u[0] = *reinterpret_cast<const u32x4*>(p);
  t.u[1] = *reinterpret_cast<const u32x4*>(p + 16);
  return t.v;
}

// ---- D-tile stores --------------------------------------------------------
// C/D layout: VGPR r -> row m0+r (lanes 0-15) or m0+8+r (lanes 16-31),
// col = n0 + (lane&15).

__device__ __forceinline__ void store_d_f32(f32x8 d, float* dst, int ld,
                                            int m0, int n0, float scale) {
  int lane = threadIdx.x & 31;
  int col  = n0 + (lane & 15);
  int rb   = m0 + ((lane >> 4) << 3);
#pragma unroll
  for (int r = 0; r < 8; ++r)
    dst[(size_t)(rb + r) * ld + col] = d[r] * scale;
}

__device__ __forceinline__ void store_d_bf16(f32x8 d, __bf16* dst, int ld,
                                             int m0, int n0) {
  int lane = threadIdx.x & 31;
  int col  = n0 + (lane & 15);
  int rb   = m0 + ((lane >> 4) << 3);
#pragma unroll
  for (int r = 0; r < 8; ++r)
    dst[(size_t)(rb + r) * ld + col] = (__bf16)d[r];
}

// transposed bf16 store: dstT[n][m] = D[m][n]; 16B contiguous per lane
__device__ __forceinline__ void store_dT_bf16(f32x8 d, __bf16* dstT, int ld,
                                              int m0, int n0) {
  int lane = threadIdx.x & 31;
  int n  = n0 + (lane & 15);
  int mb = m0 + ((lane >> 4) << 3);
  union { u32x4 u; __bf16 a[8]; } t;
#pragma unroll
  for (int r = 0; r < 8; ++r) t.a[r] = (__bf16)d[r];
  *reinterpret_cast<u32x4*>(dstT + (size_t)n * ld + mb) = t.u;
}

// ---- async 16B global->LDS staging (ASYNCcnt path on gfx1250) -------------
// GLOBAL_LOAD_ASYNC_TO_LDS_B128: VDST = VGPR with LDS byte offset (generic
// shared pointer low 32 bits == LDS offset per the aperture mapping),
// VADDR = 64-bit global address, no SADDR ("off").
__device__ __forceinline__ void stage16B(const __bf16* g, __bf16* l) {
#if HAVE_ASYNC_LDS
  unsigned lds_off = (unsigned)(unsigned long long)l;
  asm volatile("global_load_async_to_lds_b128 %0, %1, off"
               :: "v"(lds_off), "v"(g)
               : "memory");
#else
  *reinterpret_cast<u32x4*>(l) = *reinterpret_cast<const u32x4*>(g);
#endif
}
__device__ __forceinline__ void stage_wait() {
#if HAVE_ASYNC_LDS
#if __has_builtin(__builtin_amdgcn_s_wait_asynccnt)
  __builtin_amdgcn_s_wait_asynccnt(0);
#else
  asm volatile("s_wait_asynccnt 0x0" ::: "memory");
#endif
#endif
}

// ---------------------------------------------------------------------------
// Kernel 1: convert x [B,C,N] f32 -> xT [B,N,C] bf16 (transposed for B-operand)
// ---------------------------------------------------------------------------
__global__ void k_convert_x(const float* __restrict__ x, __bf16* __restrict__ xT) {
  int idx = blockIdx.x * blockDim.x + threadIdx.x;
  if (idx >= Bn * Cc * Nn) return;
  int n = idx % Nn;
  int c = (idx / Nn) % Cc;
  int b = idx / (Nn * Cc);
  xT[(size_t)b * Nn * Cc + (size_t)n * Cc + c] = (__bf16)x[idx];
}

// ---------------------------------------------------------------------------
// Kernel 2: convert the six 256x256 weights to bf16 (slot = blockIdx.y)
// ---------------------------------------------------------------------------
__global__ void k_convert_w(const float* w0, const float* w1, const float* w2,
                            const float* w3, const float* w4, const float* w5,
                            __bf16* __restrict__ Wb) {
  const float* srcs[6] = {w0, w1, w2, w3, w4, w5};
  const float* src = srcs[blockIdx.y];
  int idx = blockIdx.x * blockDim.x + threadIdx.x;
  if (idx >= Cc * Cc) return;
  Wb[(size_t)blockIdx.y * Cc * Cc + idx] = (__bf16)src[idx];
}

// ---------------------------------------------------------------------------
// Kernel 3: projections Y = W * X  for p in {q1,k1,q2,k2,v}, both batches.
// 2x2 blocking: each wave produces a 32(o) x 32(n) block.
// q/k stored transposed [N,C]; v stored [C,N].
// grid (9, 8, 10): x->n pair-tiles (8 waves/block), y->o pair-tiles, z=b*5+p
// ---------------------------------------------------------------------------
__global__ void k_proj(const __bf16* __restrict__ Wb, const __bf16* __restrict__ xT,
                       __bf16* q1T, __bf16* k1T, __bf16* q2T, __bf16* k2T,
                       __bf16* vM) {
  int wave = threadIdx.x >> 5;
  int n0   = (blockIdx.x * 8 + wave) * 32;   // 0..2272
  int o0   = blockIdx.y * 32;                // 0..224
  int b    = blockIdx.z / 5;
  int p    = blockIdx.z % 5;

  const __bf16* W = Wb + (size_t)p * Cc * Cc;        // [Cout,Cin], A operand
  const __bf16* X = xT + (size_t)b * Nn * Cc;        // [N,C], B operand

  f32x8 z = {0.f, 0.f, 0.f, 0.f, 0.f, 0.f, 0.f, 0.f};
  f32x8 acc00 = z, acc01 = z, acc10 = z, acc11 = z;
#pragma unroll
  for (int k0 = 0; k0 < Cc; k0 += 32) {
    bf16x16 a0 = load_tile(W, Cc, o0,      k0);
    bf16x16 a1 = load_tile(W, Cc, o0 + 16, k0);
    bf16x16 b0 = load_tile(X, Cc, n0,      k0);
    bf16x16 b1 = load_tile(X, Cc, n0 + 16, k0);
    acc00 = wmma_bf16(a0, b0, acc00);
    acc01 = wmma_bf16(a0, b1, acc01);
    acc10 = wmma_bf16(a1, b0, acc10);
    acc11 = wmma_bf16(a1, b1, acc11);
  }

  if (p == 4) {                                       // v -> [C,N]
    __bf16* dst = vM + (size_t)b * Cc * Nn;
    store_d_bf16(acc00, dst, Nn, o0,      n0);
    store_d_bf16(acc01, dst, Nn, o0,      n0 + 16);
    store_d_bf16(acc10, dst, Nn, o0 + 16, n0);
    store_d_bf16(acc11, dst, Nn, o0 + 16, n0 + 16);
  } else {
    __bf16* base = (p == 0) ? q1T : (p == 1) ? k1T : (p == 2) ? q2T : k2T;
    __bf16* dstT = base + (size_t)b * Nn * Cc;        // [N,C]
    store_dT_bf16(acc00, dstT, Cc, o0,      n0);
    store_dT_bf16(acc01, dstT, Cc, o0,      n0 + 16);
    store_dT_bf16(acc10, dstT, Cc, o0 + 16, n0);
    store_dT_bf16(acc11, dstT, Cc, o0 + 16, n0 + 16);
  }
}

// ---------------------------------------------------------------------------
// Kernel 4 (per b,h): scores S[n,m] = scale * sum_c q[n,c]*k[m,c]
// 2x2 blocking: wave produces 32(n) x 32(m). d=32 -> single K step.
// grid (9, 72, 2): x->m pair-tiles (8 waves), y->n pair-tiles, z: pair 1/2
// ---------------------------------------------------------------------------
__global__ void k_scores(const __bf16* q1p, const __bf16* k1p, float* P1,
                         const __bf16* q2p, const __bf16* k2p, float* P2) {
  int wave = threadIdx.x >> 5;
  int m0   = (blockIdx.x * 8 + wave) * 32;
  int n0   = blockIdx.y * 32;
  const __bf16* q = blockIdx.z ? q2p : q1p;
  const __bf16* k = blockIdx.z ? k2p : k1p;
  float*        P = blockIdx.z ? P2  : P1;

  bf16x16 a0 = load_tile(q, Cc, n0,      0);
  bf16x16 a1 = load_tile(q, Cc, n0 + 16, 0);
  bf16x16 b0 = load_tile(k, Cc, m0,      0);
  bf16x16 b1 = load_tile(k, Cc, m0 + 16, 0);
  f32x8 z = {0.f, 0.f, 0.f, 0.f, 0.f, 0.f, 0.f, 0.f};
  store_d_f32(wmma_bf16(a0, b0, z), P, Nn, n0,      m0,      SCALE);
  store_d_f32(wmma_bf16(a0, b1, z), P, Nn, n0,      m0 + 16, SCALE);
  store_d_f32(wmma_bf16(a1, b0, z), P, Nn, n0 + 16, m0,      SCALE);
  store_d_f32(wmma_bf16(a1, b1, z), P, Nn, n0 + 16, m0 + 16, SCALE);
}

// ---------------------------------------------------------------------------
// Kernel 5 (per b,h): Diff[n,m] = softmax(P1)[n,m] - softmax(P2)[n,m] (bf16)
// one 256-thread block per row n
// ---------------------------------------------------------------------------
__device__ __forceinline__ float blk_reduce_max(float v, float* s) {
  int t = threadIdx.x;
  s[t] = v; __syncthreads();
  for (int off = 128; off > 0; off >>= 1) {
    if (t < off) s[t] = fmaxf(s[t], s[t + off]);
    __syncthreads();
  }
  float r = s[0]; __syncthreads(); return r;
}
__device__ __forceinline__ float blk_reduce_sum(float v, float* s) {
  int t = threadIdx.x;
  s[t] = v; __syncthreads();
  for (int off = 128; off > 0; off >>= 1) {
    if (t < off) s[t] += s[t + off];
    __syncthreads();
  }
  float r = s[0]; __syncthreads(); return r;
}

__global__ void k_softmax_diff(const float* __restrict__ P1,
                               const float* __restrict__ P2,
                               __bf16* __restrict__ Diff) {
  __shared__ float sred[256];
  int n = blockIdx.x;
  const float* r1 = P1 + (size_t)n * Nn;
  const float* r2 = P2 + (size_t)n * Nn;

  float m1 = -3.4e38f, m2 = -3.4e38f;
  for (int m = threadIdx.x; m < Nn; m += 256) {
    m1 = fmaxf(m1, r1[m]);
    m2 = fmaxf(m2, r2[m]);
  }
  m1 = blk_reduce_max(m1, sred);
  m2 = blk_reduce_max(m2, sred);

  float s1 = 0.f, s2 = 0.f;
  for (int m = threadIdx.x; m < Nn; m += 256) {
    s1 += __expf(r1[m] - m1);
    s2 += __expf(r2[m] - m2);
  }
  s1 = blk_reduce_sum(s1, sred);
  s2 = blk_reduce_sum(s2, sred);
  float i1 = 1.f / s1, i2 = 1.f / s2;

  __bf16* drow = Diff + (size_t)n * Nn;
  for (int m = threadIdx.x; m < Nn; m += 256)
    drow[m] = (__bf16)(__expf(r1[m] - m1) * i1 - __expf(r2[m] - m2) * i2);
}

// ---------------------------------------------------------------------------
// Kernel 6 (per b,h): outT[n,c] = sum_m Diff[n,m] * v[c,m]
// Each wave: 32(n) x 32(c = full head). The 32x32 v K-slab is shared by all
// 8 waves -> staged once per block into LDS (async DMA on gfx1250).
// grid (9): 8 waves/block, 256 n-rows per block.
// ---------------------------------------------------------------------------
__global__ void k_pv(const __bf16* __restrict__ Diff,
                     const __bf16* __restrict__ vh,    // v + (b*C + h*32)*N
                     __bf16* __restrict__ outTh) {     // outT + b*N*C + h*32
  __shared__ __bf16 vtile[32 * 32];                    // [c][k], 2 KB
  int wave = threadIdx.x >> 5;
  int n0   = (blockIdx.x * 8 + wave) * 32;

  f32x8 z = {0.f, 0.f, 0.f, 0.f, 0.f, 0.f, 0.f, 0.f};
  f32x8 acc00 = z, acc01 = z, acc10 = z, acc11 = z;

  for (int m0 = 0; m0 < Nn; m0 += 32) {
    __syncthreads();                       // prior reads of vtile complete
    if (threadIdx.x < 128) {               // waves 0..3: 128 x 16B = 2 KB
      int c  = threadIdx.x >> 2;
      int ch = (threadIdx.x & 3) * 8;
      stage16B(vh + (size_t)c * Nn + m0 + ch, vtile + c * 32 + ch);
      stage_wait();                        // per-wave ASYNCcnt drain
    }
    __syncthreads();                       // staged slab visible to all waves

    bf16x16 b0 = load_tile(vtile, 32, 0,  0);   // ds_load_b128 x2
    bf16x16 b1 = load_tile(vtile, 32, 16, 0);
    bf16x16 a0 = load_tile(Diff, Nn, n0,      m0);
    bf16x16 a1 = load_tile(Diff, Nn, n0 + 16, m0);
    acc00 = wmma_bf16(a0, b0, acc00);
    acc01 = wmma_bf16(a0, b1, acc01);
    acc10 = wmma_bf16(a1, b0, acc10);
    acc11 = wmma_bf16(a1, b1, acc11);
  }

  store_d_bf16(acc00, outTh, Cc, n0,      0);   // outT[n, h*32 + c]
  store_d_bf16(acc01, outTh, Cc, n0,      16);
  store_d_bf16(acc10, outTh, Cc, n0 + 16, 0);
  store_d_bf16(acc11, outTh, Cc, n0 + 16, 16);
}

// ---------------------------------------------------------------------------
// Kernel 7: final Y[b,o,n] = sum_c Wu[o,c]*outT[b,n,c] + bu[o] + x[b,o,n]
// 2x2 blocking. grid (9, 8, 2): x->n pair-tiles, y->o pair-tiles, z->batch
// ---------------------------------------------------------------------------
__device__ __forceinline__ void store_final(f32x8 d, float* y,
                                            const float* bu, const float* x,
                                            int b, int o0, int n0) {
  int lane = threadIdx.x & 31;
  int col  = n0 + (lane & 15);
  int rb   = o0 + ((lane >> 4) << 3);
#pragma unroll
  for (int r = 0; r < 8; ++r) {
    int o = rb + r;
    size_t off = ((size_t)b * Cc + o) * Nn + col;
    y[off] = d[r] + bu[o] + x[off];
  }
}

__global__ void k_final(const __bf16* __restrict__ Wub,
                        const __bf16* __restrict__ outT,
                        const float* __restrict__ bu,
                        const float* __restrict__ x,
                        float* __restrict__ y) {
  int wave = threadIdx.x >> 5;
  int n0   = (blockIdx.x * 8 + wave) * 32;
  int o0   = blockIdx.y * 32;
  int b    = blockIdx.z;

  const __bf16* Ot = outT + (size_t)b * Nn * Cc;
  f32x8 z = {0.f, 0.f, 0.f, 0.f, 0.f, 0.f, 0.f, 0.f};
  f32x8 acc00 = z, acc01 = z, acc10 = z, acc11 = z;
#pragma unroll
  for (int k0 = 0; k0 < Cc; k0 += 32) {
    bf16x16 a0 = load_tile(Wub, Cc, o0,      k0);
    bf16x16 a1 = load_tile(Wub, Cc, o0 + 16, k0);
    bf16x16 b0 = load_tile(Ot,  Cc, n0,      k0);
    bf16x16 b1 = load_tile(Ot,  Cc, n0 + 16, k0);
    acc00 = wmma_bf16(a0, b0, acc00);
    acc01 = wmma_bf16(a0, b1, acc01);
    acc10 = wmma_bf16(a1, b0, acc10);
    acc11 = wmma_bf16(a1, b1, acc11);
  }
  store_final(acc00, y, bu, x, b, o0,      n0);
  store_final(acc01, y, bu, x, b, o0,      n0 + 16);
  store_final(acc10, y, bu, x, b, o0 + 16, n0);
  store_final(acc11, y, bu, x, b, o0 + 16, n0 + 16);
}

// ---------------------------------------------------------------------------
// Host side
// ---------------------------------------------------------------------------
extern "C" void kernel_launch(void* const* d_in, const int* in_sizes, int n_in,
                              void* d_out, int out_size, void* d_ws, size_t ws_size,
                              hipStream_t stream) {
  const float* x   = (const float*)d_in[0];
  const float* Wq1 = (const float*)d_in[1];
  const float* Wk1 = (const float*)d_in[2];
  const float* Wq2 = (const float*)d_in[3];
  const float* Wk2 = (const float*)d_in[4];
  const float* Wv  = (const float*)d_in[5];
  const float* Wu  = (const float*)d_in[6];
  const float* bu  = (const float*)d_in[7];
  float* y = (float*)d_out;

  // ---- workspace carve-out (~70 MB; P1/P2/Diff reused across (b,h) so they
  // stay resident in the 192 MB L2 between score/softmax/pv kernels) ----
  char* w = (char*)d_ws;
  size_t off = 0;
  auto carve = [&](size_t bytes) -> char* {
    char* p = w + off;
    off = (off + bytes + 255) & ~(size_t)255;
    return p;
  };
  __bf16* xT   = (__bf16*)carve((size_t)Bn * Nn * Cc * 2);
  __bf16* Wb   = (__bf16*)carve((size_t)6 * Cc * Cc * 2);
  __bf16* q1T  = (__bf16*)carve((size_t)Bn * Nn * Cc * 2);
  __bf16* k1T  = (__bf16*)carve((size_t)Bn * Nn * Cc * 2);
  __bf16* q2T  = (__bf16*)carve((size_t)Bn * Nn * Cc * 2);
  __bf16* k2T  = (__bf16*)carve((size_t)Bn * Nn * Cc * 2);
  __bf16* vM   = (__bf16*)carve((size_t)Bn * Cc * Nn * 2);
  __bf16* outT = (__bf16*)carve((size_t)Bn * Nn * Cc * 2);
  float*  P1   = (float*) carve((size_t)Nn * Nn * 4);
  float*  P2   = (float*) carve((size_t)Nn * Nn * 4);
  __bf16* Diff = (__bf16*)carve((size_t)Nn * Nn * 2);
  (void)ws_size; (void)in_sizes; (void)n_in; (void)out_size;

  // 1) conversions
  {
    int tot = Bn * Cc * Nn;
    k_convert_x<<<dim3((tot + 255) / 256), dim3(256), 0, stream>>>(x, xT);
    k_convert_w<<<dim3((Cc * Cc + 255) / 256, 6), dim3(256), 0, stream>>>(
        Wq1, Wk1, Wq2, Wk2, Wv, Wu, Wb);
  }

  // 2) five projections, both batches (n pair-tiles 72 -> 9 blocks of 8 waves)
  k_proj<<<dim3(9, Cc / 32, Bn * 5), dim3(256), 0, stream>>>(
      Wb, xT, q1T, k1T, q2T, k2T, vM);

  // 3) attention per (batch, head)
  for (int b = 0; b < Bn; ++b) {
    for (int h = 0; h < Hh; ++h) {
      const __bf16* q1p = q1T + (size_t)b * Nn * Cc + h * Dd;
      const __bf16* k1p = k1T + (size_t)b * Nn * Cc + h * Dd;
      const __bf16* q2p = q2T + (size_t)b * Nn * Cc + h * Dd;
      const __bf16* k2p = k2T + (size_t)b * Nn * Cc + h * Dd;
      const __bf16* vh  = vM  + ((size_t)b * Cc + h * Dd) * Nn;
      __bf16* oh        = outT + (size_t)b * Nn * Cc + h * Dd;

      k_scores<<<dim3(9, NT / 2, 2), dim3(256), 0, stream>>>(
          q1p, k1p, P1, q2p, k2p, P2);
      k_softmax_diff<<<dim3(Nn), dim3(256), 0, stream>>>(P1, P2, Diff);
      k_pv<<<dim3(9), dim3(256), 0, stream>>>(Diff, vh, oh);
    }
  }

  // 4) output projection + bias + residual
  k_final<<<dim3(9, Cc / 32, Bn), dim3(256), 0, stream>>>(
      Wb + (size_t)5 * Cc * Cc, outT, bu, x, y);
}
